// TransformerBlock_18313740550366
// MI455X (gfx1250) — matrix operand order repaired and checked by reading, not compiled, not run on hardware
//
#include <hip/hip_runtime.h>

// ---------------------------------------------------------------------------
// CDNA5 (gfx1250, wave32) transformer block:
//   bf16 WMMA GEMMs + flash attention + fused epilogues,
//   async global->LDS staging (ASYNCcnt) for non-transposed tiles.
// ---------------------------------------------------------------------------

typedef __attribute__((ext_vector_type(16))) __bf16 v16bf;
typedef __attribute__((ext_vector_type(8)))  float  v8f;

constexpr int kB  = 2;
constexpr int kS  = 2048;
constexpr int kD  = 1024;
constexpr int kH  = 16;
constexpr int kHD = 64;
constexpr int kF  = 4096;
constexpr int kM  = kB * kS;   // 4096 token rows

union F8    { v8f   v; float    f[8]; };
union BFRAG { v16bf v; unsigned u[8]; };

__device__ __forceinline__ unsigned short f32_to_bf16(float f) {
    unsigned u = __builtin_bit_cast(unsigned, f);
    u += 0x7FFFu + ((u >> 16) & 1u);            // round-to-nearest-even
    return (unsigned short)(u >> 16);
}

// ---- CDNA5 async global->LDS copy (16B per lane), tracked by ASYNCcnt ------
__device__ __forceinline__ void async_copy_b128(const void* gbase,
                                                unsigned gOffBytes,
                                                unsigned ldsOffBytes) {
    // GVS mode: mem_addr = SADDR(64b) + VADDR(32b); VDST vgpr = LDS byte addr.
    asm volatile("global_load_async_to_lds_b128 %0, %1, %2"
                 :
                 : "v"(ldsOffBytes), "v"(gOffBytes), "s"(gbase)
                 : "memory");
}
__device__ __forceinline__ void wait_async0() {
    asm volatile("s_wait_asynccnt 0" ::: "memory");
}
__device__ __forceinline__ unsigned lds_off(const void* p) {
    // generic pointer to LDS: low 32 bits are the LDS byte offset
    return (unsigned)(unsigned long long)p;
}

// ---------------------------------------------------------------------------
// f32 -> bf16 conversion pass (weights converted once; reused ~4096x each)
// ---------------------------------------------------------------------------
__global__ __launch_bounds__(256)
void cvt_f32_bf16(const float* __restrict__ src, unsigned short* __restrict__ dst, int n)
{
    int i = blockIdx.x * 256 + threadIdx.x;
    if (i < n) dst[i] = f32_to_bf16(src[i]);
}

// ---------------------------------------------------------------------------
// Tiled bf16 GEMM:  C[M,N] = A[M,K] @ B[K,N] (+bias, +residual / GELU)
// Block tile 128x128, BK=32, 256 threads = 8 waves, wave tile 32x64
// (2x4 grid of 16x16 v_wmma_f32_16x16x32_bf16 accumulators).
// A tile staged with async global->LDS DMA; B tile transposed via VGPRs.
// ---------------------------------------------------------------------------
constexpr int BM = 128, BN = 128, BK = 32;
constexpr int LDA = 36;   // ushort stride of As row  (pad vs bank conflicts)
constexpr int LDB = 36;   // ushort stride of Bs row  (Bs is [BN][BK], K-contig)

enum { EP_BF16_BIAS = 0, EP_F32_RES = 1, EP_BF16_GELU = 2 };

template <int EP>
__global__ __launch_bounds__(256)
void gemm_bf16(const unsigned short* __restrict__ A,
               const unsigned short* __restrict__ Bw,
               const float* __restrict__ bias,
               const float* __restrict__ res,
               unsigned short* __restrict__ outB,
               float* __restrict__ outF,
               int M, int N, int K)
{
    __shared__ unsigned short As[BM * LDA];
    __shared__ unsigned short Bs[BN * LDB];

    const int tid     = threadIdx.x;
    const int lane    = tid & 31;
    const int wave    = tid >> 5;       // 0..7
    const int wRow    = wave >> 1;      // 0..3  -> 32-row strip
    const int wCol    = wave & 1;       // 0..1  -> 64-col strip
    const int l16     = lane & 15;
    const int halfSel = lane >> 4;      // 0: lanes 0-15, 1: lanes 16-31
    const int m0      = blockIdx.y * BM;
    const int n0      = blockIdx.x * BN;

    F8 acc[2][4];
#pragma unroll
    for (int i = 0; i < 2; ++i)
#pragma unroll
        for (int j = 0; j < 4; ++j)
#pragma unroll
            for (int r = 0; r < 8; ++r) acc[i][j].f[r] = 0.0f;

    // per-thread staging geometry (invariant over k0)
    const int aRow = tid >> 1, aHf = tid & 1;
    const unsigned aLds = lds_off(&As[aRow * LDA + aHf * 16]);

    for (int k0 = 0; k0 < K; k0 += BK) {
        // ---- stage A tile 128x32 via async DMA (2 x 16B per thread) ----
        {
            unsigned gOff = (unsigned)(((size_t)(m0 + aRow) * K + k0 + aHf * 16) * 2);
            async_copy_b128(A, gOff,      aLds);
            async_copy_b128(A, gOff + 16, aLds + 16);
        }
        // ---- stage B tile 32x128, transposed into [BN][BK] (K contiguous) ----
        {
            int krow = tid >> 3;            // 0..31
            int uc   = (tid & 7) * 8;       // uint column, 8 uints = 16 cols
            const uint4* src = (const uint4*)(Bw + (size_t)(k0 + krow) * N + n0 + uc * 2);
            uint4 v0 = src[0], v1 = src[1];
            unsigned tmp[8] = { v0.x, v0.y, v0.z, v0.w, v1.x, v1.y, v1.z, v1.w };
#pragma unroll
            for (int q = 0; q < 8; ++q) {
                int n = uc * 2 + q * 2;
                Bs[(size_t)(n    ) * LDB + krow] = (unsigned short)(tmp[q] & 0xFFFFu);
                Bs[(size_t)(n + 1) * LDB + krow] = (unsigned short)(tmp[q] >> 16);
            }
        }
        wait_async0();        // this wave's async LDS writes are visible
        __syncthreads();      // all waves' staging done

        // ---- A fragments (contiguous k = 16*half + s, same order as B) ----
        BFRAG afr[2];
#pragma unroll
        for (int mt = 0; mt < 2; ++mt) {
            int m  = wRow * 32 + mt * 16 + l16;
            int kb = halfSel * 16;
#pragma unroll
            for (int j = 0; j < 8; ++j) {
                int kk = kb + 2 * j;
                afr[mt].u[j] = *(const unsigned*)&As[m * LDA + kk];
            }
        }
        // ---- B fragments + WMMA ----
#pragma unroll
        for (int nt = 0; nt < 4; ++nt) {
            BFRAG bfr;
            int n   = wCol * 64 + nt * 16 + l16;
            int kb2 = halfSel * 16;
#pragma unroll
            for (int j = 0; j < 8; ++j)
                bfr.u[j] = *(const unsigned*)&Bs[n * LDB + kb2 + 2 * j];
#pragma unroll
            for (int mt = 0; mt < 2; ++mt)
                acc[mt][nt].v = __builtin_amdgcn_wmma_f32_16x16x32_bf16(
                    false, afr[mt].v, false, bfr.v, (short)0, acc[mt][nt].v, false, false);
        }
        __syncthreads();
    }

    // ---- fused epilogue ----
#pragma unroll
    for (int nt = 0; nt < 4; ++nt) {
        int   col = n0 + wCol * 64 + nt * 16 + l16;
        float bv  = bias[col];
#pragma unroll
        for (int mt = 0; mt < 2; ++mt) {
#pragma unroll
            for (int r = 0; r < 8; ++r) {
                int    row = m0 + wRow * 32 + mt * 16 + r + halfSel * 8;
                size_t idx = (size_t)row * N + col;
                float  val = acc[mt][nt].f[r] + bv;
                if (EP == EP_BF16_BIAS) {
                    outB[idx] = f32_to_bf16(val);
                } else if (EP == EP_F32_RES) {
                    outF[idx] = val + res[idx];
                } else { // exact-erf GELU, bf16 out
                    float g = 0.5f * val * (1.0f + erff(val * 0.70710678118654752f));
                    outB[idx] = f32_to_bf16(g);
                }
            }
        }
    }
}

// ---------------------------------------------------------------------------
// Flash attention: grid (S/64 qtiles, B*H). 128 threads = 4 waves; each wave
// owns 16 query rows x full HD=64 in registers. Scores never touch HBM.
// Q/K tiles staged via async DMA; V needs a 16-bit transpose -> VGPR path.
// ---------------------------------------------------------------------------
constexpr int ALD = 72;    // ushort row stride for attention LDS tiles

__global__ __launch_bounds__(128)
void attention_kernel(const unsigned short* __restrict__ Q,
                      const unsigned short* __restrict__ Km,
                      const unsigned short* __restrict__ V,
                      const int* __restrict__ mask,
                      unsigned short* __restrict__ ctx)
{
    __shared__ unsigned short Qs[64 * ALD];  // [qrow][hd]
    __shared__ unsigned short Ks[64 * ALD];  // [kvrow][hd]
    __shared__ unsigned short Vs[64 * ALD];  // transposed: [hd][kvrow]
    __shared__ unsigned short Ps[64 * ALD];  // per-wave P strips [qrow][kv]

    const int tid     = threadIdx.x;
    const int lane    = tid & 31;
    const int wave    = tid >> 5;      // 0..3
    const int l16     = lane & 15;
    const int halfSel = lane >> 4;

    const int qt = blockIdx.x;         // 0..S/64-1
    const int bh = blockIdx.y;         // 0..B*H-1
    const int b  = bh >> 4;
    const int h  = bh & (kH - 1);
    const int q0 = qt * 64;

    const int sRow = tid >> 1, sHf = tid & 1;   // staging: 2 threads per row
    const unsigned qLds = lds_off(&Qs[sRow * ALD + sHf * 32]);
    const unsigned kLds = lds_off(&Ks[sRow * ALD + sHf * 32]);

    // ---- stage Q tile (64 x 64) once, via async DMA (4 x 16B / thread) ----
    {
        unsigned gOff = (unsigned)((((size_t)(b * kS + q0 + sRow) * kD) + h * kHD + sHf * 32) * 2);
#pragma unroll
        for (int q = 0; q < 4; ++q)
            async_copy_b128(Q, gOff + q * 16, qLds + q * 16);
    }

    F8 accO[4];
#pragma unroll
    for (int nh = 0; nh < 4; ++nh)
#pragma unroll
        for (int r = 0; r < 8; ++r) accO[nh].f[r] = 0.0f;

    float mrow[8], lrow[8];
#pragma unroll
    for (int r = 0; r < 8; ++r) { mrow[r] = -1e30f; lrow[r] = 0.0f; }

    const float scale = 0.125f;   // 1/sqrt(HD)

    for (int kv0 = 0; kv0 < kS; kv0 += 64) {
        __syncthreads();   // previous iteration done reading Ks/Vs

        // ---- stage K tile (kv x hd) via async DMA ----
        {
            unsigned gOff = (unsigned)((((size_t)(b * kS + kv0 + sRow) * kD) + h * kHD + sHf * 32) * 2);
#pragma unroll
            for (int q = 0; q < 4; ++q)
                async_copy_b128(Km, gOff + q * 16, kLds + q * 16);
        }
        // ---- stage V tile transposed: Vs[hd][kv] so K-pairs are contiguous ----
        {
            size_t g = ((size_t)(b * kS + kv0 + sRow) * kD) + h * kHD + sHf * 32;
            const uint4* src = (const uint4*)(V + g);
            uint4 v0 = src[0], v1 = src[1], v2 = src[2], v3 = src[3];
            unsigned tmp[16] = { v0.x, v0.y, v0.z, v0.w, v1.x, v1.y, v1.z, v1.w,
                                 v2.x, v2.y, v2.z, v2.w, v3.x, v3.y, v3.z, v3.w };
#pragma unroll
            for (int q = 0; q < 16; ++q) {
                int hd = sHf * 32 + q * 2;
                Vs[(size_t)(hd    ) * ALD + sRow] = (unsigned short)(tmp[q] & 0xFFFFu);
                Vs[(size_t)(hd + 1) * ALD + sRow] = (unsigned short)(tmp[q] >> 16);
            }
        }
        wait_async0();     // Q (first iter) + K copies issued by this wave done
        __syncthreads();

        // ---- S = Q @ K^T (16x64 per wave) ----
        F8 accS[4];
#pragma unroll
        for (int nt = 0; nt < 4; ++nt)
#pragma unroll
            for (int r = 0; r < 8; ++r) accS[nt].f[r] = 0.0f;

#pragma unroll
        for (int kc = 0; kc < 2; ++kc) {
            BFRAG afr;
            int m  = wave * 16 + l16;
            int kb = halfSel * 16 + kc * 32;
#pragma unroll
            for (int j = 0; j < 8; ++j) {
                int kk = kb + 2 * j;
                afr.u[j] = *(const unsigned*)&Qs[m * ALD + kk];
            }
#pragma unroll
            for (int nt = 0; nt < 4; ++nt) {
                BFRAG bfr;
                int n   = nt * 16 + l16;              // kv column
                int kb2 = halfSel * 16 + kc * 32;     // hd index (K dim)
#pragma unroll
                for (int j = 0; j < 8; ++j)
                    bfr.u[j] = *(const unsigned*)&Ks[n * ALD + kb2 + 2 * j];
                accS[nt].v = __builtin_amdgcn_wmma_f32_16x16x32_bf16(
                    false, afr.v, false, bfr.v, (short)0, accS[nt].v, false, false);
            }
        }

        // ---- scale + key mask ----
        float mb[4];
#pragma unroll
        for (int nt = 0; nt < 4; ++nt)
            mb[nt] = (mask[b * kS + kv0 + nt * 16 + l16] == 0) ? -1e9f : 0.0f;
#pragma unroll
        for (int nt = 0; nt < 4; ++nt)
#pragma unroll
            for (int r = 0; r < 8; ++r)
                accS[nt].f[r] = accS[nt].f[r] * scale + mb[nt];

        // ---- online softmax (rows live on 16-lane halves) ----
        float rmax[8];
#pragma unroll
        for (int r = 0; r < 8; ++r) {
            float m2 = fmaxf(fmaxf(accS[0].f[r], accS[1].f[r]),
                             fmaxf(accS[2].f[r], accS[3].f[r]));
#pragma unroll
            for (int off = 1; off < 16; off <<= 1)
                m2 = fmaxf(m2, __shfl_xor(m2, off, 32));
            rmax[r] = m2;
        }
        float mnew[8], alpha[8], rsum[8];
#pragma unroll
        for (int r = 0; r < 8; ++r) {
            mnew[r]  = fmaxf(mrow[r], rmax[r]);
            alpha[r] = __expf(mrow[r] - mnew[r]);
            mrow[r]  = mnew[r];
            rsum[r]  = 0.0f;
        }
#pragma unroll
        for (int nt = 0; nt < 4; ++nt)
#pragma unroll
            for (int r = 0; r < 8; ++r) {
                float p = __expf(accS[nt].f[r] - mnew[r]);
                accS[nt].f[r] = p;
                rsum[r] += p;
            }
#pragma unroll
        for (int r = 0; r < 8; ++r) {
#pragma unroll
            for (int off = 1; off < 16; off <<= 1)
                rsum[r] += __shfl_xor(rsum[r], off, 32);
            lrow[r] = lrow[r] * alpha[r] + rsum[r];
        }
#pragma unroll
        for (int nh = 0; nh < 4; ++nh)
#pragma unroll
            for (int r = 0; r < 8; ++r) accO[nh].f[r] *= alpha[r];

        // ---- P: C-layout -> A-layout via per-wave LDS strip ----
#pragma unroll
        for (int nt = 0; nt < 4; ++nt)
#pragma unroll
            for (int r = 0; r < 8; ++r) {
                int row = wave * 16 + r + halfSel * 8;
                int col = nt * 16 + l16;
                Ps[row * ALD + col] = f32_to_bf16(accS[nt].f[r]);
            }

        // ---- O += P @ V ----
#pragma unroll
        for (int kc = 0; kc < 2; ++kc) {
            BFRAG pfr;
            int m  = wave * 16 + l16;
            int kb = halfSel * 16 + kc * 32;
#pragma unroll
            for (int j = 0; j < 8; ++j) {
                int kk = kb + 2 * j;
                pfr.u[j] = *(const unsigned*)&Ps[m * ALD + kk];
            }
#pragma unroll
            for (int nh = 0; nh < 4; ++nh) {
                BFRAG vfr;
                int n   = nh * 16 + l16;              // hd column
                int kb2 = halfSel * 16 + kc * 32;     // kv index (K dim)
#pragma unroll
                for (int j = 0; j < 8; ++j)
                    vfr.u[j] = *(const unsigned*)&Vs[n * ALD + kb2 + 2 * j];
                accO[nh].v = __builtin_amdgcn_wmma_f32_16x16x32_bf16(
                    false, pfr.v, false, vfr.v, (short)0, accO[nh].v, false, false);
            }
        }
    }

    // ---- finalize: O /= l, write bf16 ctx in (B,S,H,HD) layout ----
    float inv[8];
#pragma unroll
    for (int r = 0; r < 8; ++r) inv[r] = 1.0f / lrow[r];
#pragma unroll
    for (int nh = 0; nh < 4; ++nh)
#pragma unroll
        for (int r = 0; r < 8; ++r) {
            int row = q0 + wave * 16 + r + halfSel * 8;
            int col = nh * 16 + l16;
            ctx[((size_t)(b * kS + row)) * kD + h * kHD + col] =
                f32_to_bf16(accO[nh].f[r] * inv[r]);
        }
}

// ---------------------------------------------------------------------------
// LayerNorm: one 256-thread block (8 waves) per token row; emits f32 (residual
// path) and optionally bf16 (next GEMM operand) in one pass.
// ---------------------------------------------------------------------------
__global__ __launch_bounds__(256)
void layernorm_kernel(const float* __restrict__ x,
                      const float* __restrict__ g,
                      const float* __restrict__ beta,
                      float* __restrict__ outF,
                      unsigned short* __restrict__ outB)
{
    __shared__ float rs[8], rs2[8];
    const int tid  = threadIdx.x;
    const int lane = tid & 31;
    const int wave = tid >> 5;
    const int row  = blockIdx.x;
    const float* xr = x + (size_t)row * kD;

    float s = 0.0f, s2 = 0.0f;
    for (int i = tid; i < kD; i += 256) {
        float v = xr[i];
        s += v; s2 += v * v;
    }
#pragma unroll
    for (int off = 16; off >= 1; off >>= 1) {
        s  += __shfl_xor(s,  off, 32);
        s2 += __shfl_xor(s2, off, 32);
    }
    if (lane == 0) { rs[wave] = s; rs2[wave] = s2; }
    __syncthreads();
    if (tid == 0) {
        float a = 0.0f, c = 0.0f;
        for (int i = 0; i < 8; ++i) { a += rs[i]; c += rs2[i]; }
        rs[0] = a; rs2[0] = c;
    }
    __syncthreads();
    float mean = rs[0] * (1.0f / kD);
    float var  = rs2[0] * (1.0f / kD) - mean * mean;
    float rstd = rsqrtf(var + 1e-5f);

    for (int i = tid; i < kD; i += 256) {
        float v = (xr[i] - mean) * rstd * g[i] + beta[i];
        outF[(size_t)row * kD + i] = v;
        if (outB) outB[(size_t)row * kD + i] = f32_to_bf16(v);
    }
}

// ---------------------------------------------------------------------------
// Host-side pipeline
// ---------------------------------------------------------------------------
extern "C" void kernel_launch(void* const* d_in, const int* in_sizes, int n_in,
                              void* d_out, int out_size, void* d_ws, size_t ws_size,
                              hipStream_t stream)
{
    const float* x    = (const float*)d_in[0];
    const int*   mask = (const int*)  d_in[1];
    const float* Wq = (const float*)d_in[2];   const float* bq  = (const float*)d_in[3];
    const float* Wk = (const float*)d_in[4];   const float* bk  = (const float*)d_in[5];
    const float* Wv = (const float*)d_in[6];   const float* bv  = (const float*)d_in[7];
    const float* Wo = (const float*)d_in[8];   const float* bo  = (const float*)d_in[9];
    const float* g1 = (const float*)d_in[10];  const float* be1 = (const float*)d_in[11];
    const float* W1 = (const float*)d_in[12];  const float* b1  = (const float*)d_in[13];
    const float* W2 = (const float*)d_in[14];  const float* b2  = (const float*)d_in[15];
    const float* g2 = (const float*)d_in[16];  const float* be2 = (const float*)d_in[17];

    char*  ws  = (char*)d_ws;
    size_t off = 0;
    auto alloc = [&](size_t bytes) -> void* {
        void* p = ws + off;
        off = (off + bytes + 255) & ~(size_t)255;
        return p;
    };

    const size_t MD = (size_t)kM * kD;          // 4M elements
    unsigned short* xb   = (unsigned short*)alloc(MD * 2);
    unsigned short* wqb  = (unsigned short*)alloc((size_t)kD * kD * 2);
    unsigned short* wkb  = (unsigned short*)alloc((size_t)kD * kD * 2);
    unsigned short* wvb  = (unsigned short*)alloc((size_t)kD * kD * 2);
    unsigned short* wob  = (unsigned short*)alloc((size_t)kD * kD * 2);
    unsigned short* w1b  = (unsigned short*)alloc((size_t)kD * kF * 2);
    unsigned short* w2b  = (unsigned short*)alloc((size_t)kF * kD * 2);
    unsigned short* qb   = (unsigned short*)alloc(MD * 2);
    unsigned short* kb_  = (unsigned short*)alloc(MD * 2);
    unsigned short* vb_  = (unsigned short*)alloc(MD * 2);
    unsigned short* ctxb = (unsigned short*)alloc(MD * 2);
    float*          y1   = (float*)alloc(MD * 4);   // x + attn_out
    float*          n1f  = (float*)alloc(MD * 4);   // layernorm1 f32
    unsigned short* n1b  = (unsigned short*)alloc(MD * 2);
    unsigned short* hb   = (unsigned short*)alloc((size_t)kM * kF * 2);  // gelu(...)
    float*          y2   = (float*)alloc(MD * 4);   // n1 + mlp_out
    (void)ws_size; (void)in_sizes; (void)n_in; (void)out_size;

    // --- precision conversion (one pass; weights reused ~4096x each) ---
    auto cvt = [&](const float* s, unsigned short* d, size_t n) {
        cvt_f32_bf16<<<dim3((unsigned)((n + 255) / 256)), dim3(256), 0, stream>>>(s, d, (int)n);
    };
    cvt(x,  xb,  MD);
    cvt(Wq, wqb, (size_t)kD * kD);
    cvt(Wk, wkb, (size_t)kD * kD);
    cvt(Wv, wvb, (size_t)kD * kD);
    cvt(Wo, wob, (size_t)kD * kD);
    cvt(W1, w1b, (size_t)kD * kF);
    cvt(W2, w2b, (size_t)kF * kD);

    dim3 blk(256);
    dim3 gridDD(kD / BN, kM / BM);   // (8, 32)
    dim3 gridDF(kF / BN, kM / BM);   // (32, 32)

    // --- QKV projections (bias fused, bf16 out) ---
    gemm_bf16<EP_BF16_BIAS><<<gridDD, blk, 0, stream>>>(xb, wqb, bq, nullptr, qb,  nullptr, kM, kD, kD);
    gemm_bf16<EP_BF16_BIAS><<<gridDD, blk, 0, stream>>>(xb, wkb, bk, nullptr, kb_, nullptr, kM, kD, kD);
    gemm_bf16<EP_BF16_BIAS><<<gridDD, blk, 0, stream>>>(xb, wvb, bv, nullptr, vb_, nullptr, kM, kD, kD);

    // --- flash attention (scores never hit HBM) ---
    dim3 attnGrid(kS / 64, kB * kH);  // (32, 32)
    attention_kernel<<<attnGrid, dim3(128), 0, stream>>>(qb, kb_, vb_, mask, ctxb);

    // --- output projection + residual (f32) ---
    gemm_bf16<EP_F32_RES><<<gridDD, blk, 0, stream>>>(ctxb, wob, bo, x, nullptr, y1, kM, kD, kD);

    // --- LN1 (f32 + bf16 outs) ---
    layernorm_kernel<<<dim3(kM), blk, 0, stream>>>(y1, g1, be1, n1f, n1b);

    // --- MLP up-proj + exact GELU (bf16 out) ---
    gemm_bf16<EP_BF16_GELU><<<gridDF, blk, 0, stream>>>(n1b, w1b, b1, nullptr, hb, nullptr, kM, kF, kD);

    // --- MLP down-proj + residual with n1 (f32) ---
    gemm_bf16<EP_F32_RES><<<gridDD, blk, 0, stream>>>(hb, w2b, b2, n1f, nullptr, y2, kM, kD, kF);

    // --- LN2 -> final output (f32) ---
    layernorm_kernel<<<dim3(kM), blk, 0, stream>>>(y2, g2, be2, (float*)d_out, nullptr);
}
